// RefineVitBlock_24644522344901
// MI455X (gfx1250) — matrix operand order repaired and checked by reading, not compiled, not run on hardware
//
#include <hip/hip_runtime.h>
#include <math.h>

// ---------------------------------------------------------------------------
// RefineVitBlock for MI455X (gfx1250, wave32, WMMA bf16 16x16x32, f32 accum)
//
// Roofline: ~70 GFLOP matmul work, ~280 MB compulsory HBM traffic.
//   HBM floor  : 280 MB / 23.3 TB/s  ~ 12 us
//   bf16 WMMA  : ~70 GFLOP at multi-PFLOP/s ~ 25-30 us  -> compute bound.
// All WMMA fragments are contiguous b128 vector loads (A: row-major 2x16B,
// B: transposed source 32B/lane). GEMM kernels: per-wave fixed m-tile with
// A-fragments preloaded once, dual n-tile accumulator chains for load/WMMA
// overlap. Flash softmax: row max via lane shuffles; row SUM via an extra
// WMMA against an all-ones B (every lane receives its row sum in D-layout),
// eliminating half the ds_bpermute traffic.
// ---------------------------------------------------------------------------

typedef __bf16 bf16;
typedef __attribute__((ext_vector_type(16))) __bf16 bf16x16;
typedef __attribute__((ext_vector_type(8)))  __bf16 bf16x8;
typedef __attribute__((ext_vector_type(8)))  float  f32x8;

#define HEADS   4
#define CH      64
#define Cc      256
#define Hh      256
#define Ww      256
#define Bb      2
#define WINSZ   64
#define NWIN    1024
#define NWF     307
#define NWSEL   (Bb * NWF)      // 614
#define HW1     1024            // 32*32 sr tokens per batch
#define L_GLOB  1024
#define NEG_INF (-3.0e38f)

__device__ __forceinline__ int lane_id() { return threadIdx.x & 31; }
__device__ __forceinline__ int wave_id() { return threadIdx.x >> 5; }

__device__ __forceinline__ f32x8 wmma_bf16(bf16x16 a, bf16x16 b, f32x8 c) {
  return __builtin_amdgcn_wmma_f32_16x16x32_bf16(false, a, false, b,
                                                 (short)0, c, false, false);
}

// ---- vectorized fragment loaders per CDNA5 ISA 7.12.2 ---------------------
// A 16x32 bf16, row-major src (ld in elements). Per lane (r=l&15, h=l>>4):
// elements k = 8h..8h+7 (vgprs 0-3) and k = 16+8h..16+8h+7 (vgprs 4-7).
__device__ __forceinline__ bf16x16 load_a_vec(const bf16* src, int ld) {
  const int l = lane_id(), half = l >> 4, r = l & 15;
  const bf16* p = src + (size_t)r * ld + 8 * half;
  bf16x8 lo = *(const bf16x8*)p;
  bf16x8 hi = *(const bf16x8*)(p + 16);
  return __builtin_shufflevector(lo, hi, 0, 1, 2, 3, 4, 5, 6, 7,
                                 8, 9, 10, 11, 12, 13, 14, 15);
}
// B 32x16 bf16 from transposed source: B(k,n) = src[n*ld + k].
// Per lane (n=l&15, h=l>>4): elements k = 8h..8h+7 and k = 16+8h..16+8h+7.
__device__ __forceinline__ bf16x16 load_bt_vec(const bf16* src, int ld) {
  const int l = lane_id(), half = l >> 4, n = l & 15;
  const bf16* p = src + (size_t)n * ld + 8 * half;
  bf16x8 lo = *(const bf16x8*)p;
  bf16x8 hi = *(const bf16x8*)(p + 16);
  return __builtin_shufflevector(lo, hi, 0, 1, 2, 3, 4, 5, 6, 7,
                                 8, 9, 10, 11, 12, 13, 14, 15);
}
// D f32 16x16 row-major store: VGPR i -> row i+8h, col = lane&15 (scalar).
__device__ __forceinline__ void store_d_bf16(bf16* dst, int ld, f32x8 c) {
  const int l = lane_id(), half = l >> 4, n = l & 15;
#pragma unroll
  for (int i = 0; i < 8; ++i) dst[(size_t)(i + 8 * half) * ld + n] = (bf16)c[i];
}
// D transposed store: dst[n*ld + m]; per lane 8 contiguous bf16 -> b128 store.
__device__ __forceinline__ void store_dt_bf16(bf16* dst, int ld, f32x8 c) {
  const int l = lane_id(), half = l >> 4, n = l & 15;
  bf16x8 v;
#pragma unroll
  for (int i = 0; i < 8; ++i) v[i] = (bf16)c[i];
  *(bf16x8*)(dst + (size_t)n * ld + 8 * half) = v;
}

// ---------------------------------------------------------------------------
// Flash-attention unit: 16 query rows x 64-ch head slice vs L keys.
// Q row-major [16 x 64] (ld ldq). K row-major [L x ldk] head slice. Vt
// transposed [64ch x ldv] head slice. Online softmax; per-(VGPR i, lane-half)
// row stats replicated across the 16 lanes of the half (matches the D-layout
// row mapping). P re-enters A-layout via a per-wave LDS bounce. The softmax
// denominator is accumulated as a 5th WMMA output (P @ ones -> row sums
// delivered to every lane in D-layout), rescaled by the same alpha as O.
// ---------------------------------------------------------------------------
__device__ void flash_unit(const bf16* Q, int ldq,
                           const bf16* K, int ldk,
                           const bf16* Vt, int ldv,
                           int L, float scale,
                           bf16* psc, float* accOut, int ldacc) {
  bf16x16 qa0 = load_a_vec(Q, ldq);        // ch 0..31
  bf16x16 qa1 = load_a_vec(Q + 32, ldq);   // ch 32..63
  bf16x16 ones;
#pragma unroll
  for (int j = 0; j < 16; ++j) ones[j] = (bf16)1.0f;
  f32x8 o0 = {}, o1 = {}, o2 = {}, o3 = {}, osum = {};
  float mrow[8];
#pragma unroll
  for (int i = 0; i < 8; ++i) mrow[i] = NEG_INF;
  const int l = lane_id(), half = l >> 4, n = l & 15;

  for (int kt = 0; kt < L; kt += 32) {
    // issue V loads early; they are consumed only after the softmax chain
    bf16x16 vb0 = load_bt_vec(Vt + (size_t)(0)  * ldv + kt, ldv);
    bf16x16 vb1 = load_bt_vec(Vt + (size_t)(16) * ldv + kt, ldv);
    bf16x16 vb2 = load_bt_vec(Vt + (size_t)(32) * ldv + kt, ldv);
    bf16x16 vb3 = load_bt_vec(Vt + (size_t)(48) * ldv + kt, ldv);
    f32x8 s0 = {}, s1 = {};
    s0 = wmma_bf16(qa0, load_bt_vec(K + (size_t)kt * ldk, ldk), s0);
    s1 = wmma_bf16(qa0, load_bt_vec(K + (size_t)(kt + 16) * ldk, ldk), s1);
    s0 = wmma_bf16(qa1, load_bt_vec(K + (size_t)kt * ldk + 32, ldk), s0);
    s1 = wmma_bf16(qa1, load_bt_vec(K + (size_t)(kt + 16) * ldk + 32, ldk), s1);
#pragma unroll
    for (int i = 0; i < 8; ++i) {
      float a0 = s0[i] * scale, a1 = s1[i] * scale;
      float t = fmaxf(a0, a1);
#pragma unroll
      for (int m = 1; m < 16; m <<= 1) t = fmaxf(t, __shfl_xor(t, m, 32));
      float mnew  = fmaxf(mrow[i], t);
      float alpha = __expf(mrow[i] - mnew);
      mrow[i] = mnew;
      o0[i] *= alpha; o1[i] *= alpha; o2[i] *= alpha; o3[i] *= alpha;
      osum[i] *= alpha;
      psc[(i + 8 * half) * 32 + n]      = (bf16)__expf(a0 - mnew);
      psc[(i + 8 * half) * 32 + 16 + n] = (bf16)__expf(a1 - mnew);
    }
    bf16x16 pa = load_a_vec(psc, 32);   // wave-local LDS RAW, DS is in-order
    o0 = wmma_bf16(pa, vb0, o0);
    o1 = wmma_bf16(pa, vb1, o1);
    o2 = wmma_bf16(pa, vb2, o2);
    o3 = wmma_bf16(pa, vb3, o3);
    osum = wmma_bf16(pa, ones, osum);   // row sums, replicated to all lanes
  }
#pragma unroll
  for (int i = 0; i < 8; ++i) {
    float inv = 1.f / osum[i];
    float* row = accOut + (size_t)(i + 8 * half) * ldacc + n;
    row[0]  += o0[i] * inv;
    row[16] += o1[i] * inv;
    row[32] += o2[i] * inv;
    row[48] += o3[i] * inv;
  }
}

// ---------------------------------------------------------------------------
// Kernels
// ---------------------------------------------------------------------------
// f32 [K x N] -> bf16 transposed [N x K]
__global__ void k_cvt_t(const float* __restrict__ src, bf16* __restrict__ dst,
                        int K, int N) {
  int i = blockIdx.x * blockDim.x + threadIdx.x;
  if (i < K * N) {
    int k = i / N, n = i - k * N;
    dst[(size_t)n * K + k] = (bf16)src[i];
  }
}

// Top-k window selection: one wave per batch, shuffle-reduce argmax, 307 iters.
__global__ void k_topk(const float* __restrict__ unc, int* __restrict__ idx) {
  __shared__ float ssc[NWIN];
  int b = blockIdx.x, lane = threadIdx.x;
  for (int i = 0; i < 32; ++i) {
    int w = i * 32 + lane;
    int y0 = (w >> 5) * 8, x0 = (w & 31) * 8;
    const float* up = unc + ((size_t)b * Hh + y0) * Ww + x0;
    float s = 0.f;
    for (int yy = 0; yy < 8; ++yy)
#pragma unroll
      for (int xx = 0; xx < 8; ++xx) s += up[yy * Ww + xx];
    ssc[w] = s;  // mean/64 is order-preserving
  }
  __syncthreads();
  for (int it = 0; it < NWF; ++it) {
    float m = NEG_INF; int mi = 0;
    for (int i = 0; i < 32; ++i) {
      float v = ssc[i * 32 + lane];
      if (v > m) { m = v; mi = i * 32 + lane; }   // strict > : stable (low idx)
    }
    for (int off = 16; off >= 1; off >>= 1) {
      float ov = __shfl_xor(m, off, 32);
      int   oi = __shfl_xor(mi, off, 32);
      if (ov > m || (ov == m && oi < mi)) { m = ov; mi = oi; }
    }
    if (lane == 0) { idx[b * NWF + it] = b * NWIN + mi; ssc[mi] = NEG_INF; }
    __syncthreads();
  }
}

// Gather selected windows: NCHW -> [win][tok][c], LDS transpose staging (32KB)
__global__ __launch_bounds__(256) void k_gather(const float* __restrict__ fm,
                                                const int* __restrict__ idx,
                                                float* __restrict__ Acc) {
  __shared__ float tile[32][Cc];
  int j = blockIdx.x;
  int g = idx[j];
  int b = g >> 10, w = g & 1023;
  int y0 = (w >> 5) * 8, x0 = (w & 31) * 8;
  for (int p = 0; p < 2; ++p) {
    int t  = threadIdx.x & 31;   // token within half
    int cq = threadIdx.x >> 5;   // 8 channel groups
    int tok = p * 32 + t;
    int y = y0 + (tok >> 3), x = x0 + (tok & 7);
    for (int c = cq; c < Cc; c += 8)
      tile[t][c] = fm[(((size_t)b * Cc + c) * Hh + y) * Ww + x];
    __syncthreads();
    int c2 = threadIdx.x;
    float* dst = Acc + (size_t)j * WINSZ * Cc + (size_t)p * 32 * Cc;
    for (int tt = 0; tt < 32; ++tt) dst[tt * Cc + c2] = tile[tt][c2];
    __syncthreads();
  }
}

// Per-window GEMM: Out[j] = bf16(Acc[j]) (64x256) @ W[:, col0:col0+256].
// WT is pre-transposed [N x 256]. Each wave owns one m-tile (A-fragments
// preloaded once) and 8 n-tiles, processed in dual-accumulator pairs.
// TRANS=1 stores the transposed [256 x 64] layout (for V).
template <int TRANS>
__global__ __launch_bounds__(256) void k_qkv_win(const float* __restrict__ Acc,
                                                 const bf16* __restrict__ WT,
                                                 int col0,
                                                 bf16* __restrict__ Out) {
  __shared__ __align__(32) bf16 Xs[WINSZ * Cc];   // 32KB
  int j = blockIdx.x;
  const float* A = Acc + (size_t)j * WINSZ * Cc;
  for (int i = threadIdx.x * 4; i < WINSZ * Cc; i += 1024) {
    float4 v = *(const float4*)(A + i);
    Xs[i] = (bf16)v.x; Xs[i + 1] = (bf16)v.y;
    Xs[i + 2] = (bf16)v.z; Xs[i + 3] = (bf16)v.w;
  }
  __syncthreads();
  const int mt = wave_id() & 3;            // wave-private m-tile
  const int nbase = (wave_id() >> 2) * 8;  // 8 n-tiles per wave
  bf16x16 afr[8];
#pragma unroll
  for (int kk = 0; kk < 8; ++kk)
    afr[kk] = load_a_vec(Xs + mt * 16 * Cc + kk * 32, Cc);
#pragma unroll
  for (int np = 0; np < 4; ++np) {
    int nt0 = nbase + np * 2, nt1 = nt0 + 1;
    const bf16* w0 = WT + (size_t)(col0 + nt0 * 16) * Cc;
    const bf16* w1 = WT + (size_t)(col0 + nt1 * 16) * Cc;
    f32x8 acc0 = {}, acc1 = {};
#pragma unroll
    for (int kk = 0; kk < 8; ++kk) {
      acc0 = wmma_bf16(afr[kk], load_bt_vec(w0 + kk * 32, Cc), acc0);
      acc1 = wmma_bf16(afr[kk], load_bt_vec(w1 + kk * 32, Cc), acc1);
    }
    if (TRANS) {
      store_dt_bf16(Out + (size_t)j * WINSZ * Cc + (size_t)(nt0 * 16) * WINSZ + mt * 16,
                    WINSZ, acc0);
      store_dt_bf16(Out + (size_t)j * WINSZ * Cc + (size_t)(nt1 * 16) * WINSZ + mt * 16,
                    WINSZ, acc1);
    } else {
      store_d_bf16(Out + (size_t)j * WINSZ * Cc + mt * 16 * Cc + nt0 * 16, Cc, acc0);
      store_d_bf16(Out + (size_t)j * WINSZ * Cc + mt * 16 * Cc + nt1 * 16, Cc, acc1);
    }
  }
}

// Attention over filtered windows (local: per-window KV; cross: per-batch KV)
__global__ __launch_bounds__(256) void k_win_attn(float* __restrict__ Acc,
                                                  const bf16* __restrict__ Qb,
                                                  const bf16* __restrict__ K,
                                                  const bf16* __restrict__ Vt,
                                                  int L, int kv_per_window) {
  __shared__ __align__(32) bf16 psc_all[8 * 16 * 32];  // 8KB, 1KB per wave
  int j = blockIdx.x;
  size_t kvoff = kv_per_window ? (size_t)j * WINSZ * Cc
                               : (size_t)(j / NWF) * (size_t)L_GLOB * Cc;
  const bf16* Kb  = K  + kvoff;            // [L x 256] row-major
  const bf16* Vtb = Vt + kvoff;            // [256 x L] transposed
  const bf16* Q   = Qb + (size_t)j * WINSZ * Cc;
  float* A = Acc + (size_t)j * WINSZ * Cc;
  bf16* psc = psc_all + wave_id() * 512;
  for (int u = wave_id(); u < 16; u += 8) {       // 4 heads x 4 row-tiles
    int h = u >> 2, mt = u & 3;
    flash_unit(Q + mt * 16 * Cc + h * CH, Cc,
               Kb + h * CH, Cc,
               Vtb + (size_t)(h * CH) * L, L,
               L, 0.125f, psc, A + mt * 16 * Cc + h * CH, Cc);
  }
}

// Depthwise 8x8 stride-8 conv: one block per (b,c) for coalesced plane reads
__global__ __launch_bounds__(256) void k_conv(const float* __restrict__ fm,
                                              const float* __restrict__ srw,
                                              const float* __restrict__ srb,
                                              float* __restrict__ Gtmp) {
  int bc = blockIdx.x;
  int b = bc >> 8, c = bc & 255;
  __shared__ float wloc[64];
  if (threadIdx.x < 64) wloc[threadIdx.x] = srw[c * 64 + threadIdx.x];
  __syncthreads();
  float bias = srb[c];
  const float* base = fm + ((size_t)b * Cc + c) * Hh * Ww;
  for (int t = threadIdx.x; t < HW1; t += 256) {
    int ty = t >> 5, tx = t & 31;
    const float* p = base + (ty * 8) * Ww + tx * 8;
    float s = bias;
#pragma unroll
    for (int i = 0; i < 8; ++i)
#pragma unroll
      for (int jx = 0; jx < 8; ++jx) s += p[i * Ww + jx] * wloc[i * 8 + jx];
    Gtmp[((size_t)b * HW1 + t) * Cc + c] = s;
  }
}

// LayerNorm over C + exact GELU, one block per sr token row
__global__ __launch_bounds__(256) void k_lngelu(const float* __restrict__ Gtmp,
                                                const float* __restrict__ g,
                                                const float* __restrict__ bta,
                                                bf16* __restrict__ Gbf) {
  int row = blockIdx.x;
  int c = threadIdx.x;
  float x = Gtmp[(size_t)row * Cc + c];
  __shared__ float red[256];
  red[c] = x; __syncthreads();
  for (int off = 128; off; off >>= 1) { if (c < off) red[c] += red[c + off]; __syncthreads(); }
  float mu = red[0] * (1.f / 256.f);
  __syncthreads();
  float d = x - mu;
  red[c] = d * d; __syncthreads();
  for (int off = 128; off; off >>= 1) { if (c < off) red[c] += red[c + off]; __syncthreads(); }
  float var = red[0] * (1.f / 256.f);
  float y = d * rsqrtf(var + 1e-5f) * g[c] + bta[c];
  float out = 0.5f * y * (1.f + erff(y * 0.70710678118654752f));
  Gbf[(size_t)row * Cc + c] = (bf16)out;
}

// Global-branch KV: [2048 x 256] @ wkv_g [256 x 512] -> Kg row-major,
// Vg transposed per batch ([256 x 1024] each batch). Wave-private m-tile
// with preloaded A-fragments; 16 n-tiles per wave in dual pairs; the
// K-vs-V store decision is wave-uniform (n-tile halves don't straddle).
__global__ __launch_bounds__(256) void k_kvg(const bf16* __restrict__ Gbf,
                                             const bf16* __restrict__ WkvT,
                                             bf16* __restrict__ Kg,
                                             bf16* __restrict__ Vgt) {
  __shared__ __align__(32) bf16 Xs[64 * Cc];     // 32KB
  int rb = blockIdx.x;             // 0..31 -> rows rb*64..
  int b = rb >> 4;                 // batch
  int tok0 = (rb & 15) * 64;       // token offset within batch
  const bf16* src = Gbf + (size_t)rb * 64 * Cc;
  for (int i = threadIdx.x * 8; i < 64 * Cc; i += 2048)
    *(bf16x8*)(Xs + i) = *(const bf16x8*)(src + i);
  __syncthreads();
  const int mt = wave_id() & 3;
  const int nbase = (wave_id() >> 2) * 16;   // 0 -> K tiles, 16 -> V tiles
  bf16x16 afr[8];
#pragma unroll
  for (int kk = 0; kk < 8; ++kk)
    afr[kk] = load_a_vec(Xs + mt * 16 * Cc + kk * 32, Cc);
#pragma unroll
  for (int np = 0; np < 8; ++np) {
    int nt0 = nbase + np * 2, nt1 = nt0 + 1;
    const bf16* w0 = WkvT + (size_t)(nt0 * 16) * Cc;
    const bf16* w1 = WkvT + (size_t)(nt1 * 16) * Cc;
    f32x8 acc0 = {}, acc1 = {};
#pragma unroll
    for (int kk = 0; kk < 8; ++kk) {
      acc0 = wmma_bf16(afr[kk], load_bt_vec(w0 + kk * 32, Cc), acc0);
      acc1 = wmma_bf16(afr[kk], load_bt_vec(w1 + kk * 32, Cc), acc1);
    }
    if (nbase == 0) {   // K -> row-major [2048 x 256]
      store_d_bf16(Kg + ((size_t)rb * 64 + mt * 16) * Cc + nt0 * 16, Cc, acc0);
      store_d_bf16(Kg + ((size_t)rb * 64 + mt * 16) * Cc + nt1 * 16, Cc, acc1);
    } else {            // V -> transposed [b][256 x 1024]
      int vc0 = nt0 * 16 - Cc, vc1 = nt1 * 16 - Cc;
      store_dt_bf16(Vgt + ((size_t)b * Cc + vc0) * L_GLOB + tok0 + mt * 16,
                    L_GLOB, acc0);
      store_dt_bf16(Vgt + ((size_t)b * Cc + vc1) * L_GLOB + tok0 + mt * 16,
                    L_GLOB, acc1);
    }
  }
}

// out = feature_map (float4 vectorized)
__global__ void k_copy(const float* __restrict__ fm, float* __restrict__ out, size_t n4) {
  size_t i = (size_t)blockIdx.x * blockDim.x + threadIdx.x;
  const float4* s = (const float4*)fm;
  float4* d = (float4*)out;
  for (; i < n4; i += (size_t)gridDim.x * blockDim.x) d[i] = s[i];
}

// Projection (Acc @ wp + bp) and scatter-add into out (unique windows -> no atomics)
__global__ __launch_bounds__(256) void k_proj(const float* __restrict__ Acc,
                                              const int* __restrict__ idx,
                                              const bf16* __restrict__ WpT,
                                              const float* __restrict__ bp,
                                              float* __restrict__ out) {
  __shared__ __align__(32) bf16 Xs[WINSZ * Cc];   // 32KB
  int j = blockIdx.x;
  int g = idx[j];
  int b = g >> 10, w = g & 1023;
  int y0 = (w >> 5) * 8, x0 = (w & 31) * 8;
  const float* A = Acc + (size_t)j * WINSZ * Cc;
  for (int i = threadIdx.x * 4; i < WINSZ * Cc; i += 1024) {
    float4 v = *(const float4*)(A + i);
    Xs[i] = (bf16)v.x; Xs[i + 1] = (bf16)v.y;
    Xs[i + 2] = (bf16)v.z; Xs[i + 3] = (bf16)v.w;
  }
  __syncthreads();
  const int l = lane_id(), half = l >> 4, n = l & 15;
  const int mt = wave_id() & 3;
  const int nbase = (wave_id() >> 2) * 8;
  bf16x16 afr[8];
#pragma unroll
  for (int kk = 0; kk < 8; ++kk)
    afr[kk] = load_a_vec(Xs + mt * 16 * Cc + kk * 32, Cc);
#pragma unroll
  for (int np = 0; np < 4; ++np) {
    int nt0 = nbase + np * 2, nt1 = nt0 + 1;
    const bf16* w0 = WpT + (size_t)(nt0 * 16) * Cc;
    const bf16* w1 = WpT + (size_t)(nt1 * 16) * Cc;
    f32x8 acc0 = {}, acc1 = {};
#pragma unroll
    for (int kk = 0; kk < 8; ++kk) {
      acc0 = wmma_bf16(afr[kk], load_bt_vec(w0 + kk * 32, Cc), acc0);
      acc1 = wmma_bf16(afr[kk], load_bt_vec(w1 + kk * 32, Cc), acc1);
    }
#pragma unroll
    for (int pp = 0; pp < 2; ++pp) {
      const f32x8& acc = pp ? acc1 : acc0;
      int c = (pp ? nt1 : nt0) * 16 + n;
      float bias = bp[c];
#pragma unroll
      for (int i = 0; i < 8; ++i) {
        int tok = mt * 16 + i + 8 * half;
        int y = y0 + (tok >> 3), x = x0 + (tok & 7);
        size_t o = (((size_t)b * Cc + c) * Hh + y) * Ww + x;
        out[o] += acc[i] + bias;
      }
    }
  }
}

// ---------------------------------------------------------------------------
extern "C" void kernel_launch(void* const* d_in, const int* in_sizes, int n_in,
                              void* d_out, int out_size, void* d_ws, size_t ws_size,
                              hipStream_t stream) {
  const float* fm    = (const float*)d_in[0];
  const float* unc   = (const float*)d_in[1];
  const float* srw   = (const float*)d_in[2];
  const float* srb   = (const float*)d_in[3];
  const float* lng   = (const float*)d_in[4];
  const float* lnb   = (const float*)d_in[5];
  const float* wq_l  = (const float*)d_in[6];
  const float* wkv_l = (const float*)d_in[7];
  const float* wq_g  = (const float*)d_in[8];
  const float* wkv_g = (const float*)d_in[9];
  const float* wp    = (const float*)d_in[10];
  const float* bp    = (const float*)d_in[11];
  float* out = (float*)d_out;

  // workspace carve-out (~108 MB total)
  char* ws = (char*)d_ws;
  size_t o = 0;
  auto take = [&](size_t nbytes) -> char* {
    char* p = ws + o;
    o += (nbytes + 255) & ~(size_t)255;
    return p;
  };
  bf16*  wq_l_t  = (bf16*)take((size_t)65536 * 2);    // [256 x 256]
  bf16*  wkv_l_t = (bf16*)take((size_t)131072 * 2);   // [512 x 256]
  bf16*  wq_g_t  = (bf16*)take((size_t)65536 * 2);    // [256 x 256]
  bf16*  wkv_g_t = (bf16*)take((size_t)131072 * 2);   // [512 x 256]
  bf16*  wp_t    = (bf16*)take((size_t)65536 * 2);    // [256 x 256]
  int*   idx     = (int*)take((size_t)NWSEL * 4);
  float* Acc     = (float*)take((size_t)NWSEL * WINSZ * Cc * 4);
  float* Gtmp    = (float*)take((size_t)Bb * HW1 * Cc * 4);
  bf16*  Gbf     = (bf16*)take((size_t)Bb * HW1 * Cc * 2);
  bf16*  Kg      = (bf16*)take((size_t)Bb * HW1 * Cc * 2);  // row-major
  bf16*  Vgt     = (bf16*)take((size_t)Bb * HW1 * Cc * 2);  // [b][256 x 1024]
  bf16*  Qb      = (bf16*)take((size_t)NWSEL * WINSZ * Cc * 2); // row-major
  bf16*  Kb      = (bf16*)take((size_t)NWSEL * WINSZ * Cc * 2); // row-major
  bf16*  Vbt     = (bf16*)take((size_t)NWSEL * WINSZ * Cc * 2); // [j][256 x 64]

  // weights -> bf16, transposed to [N x K]
  k_cvt_t<<<256, 256, 0, stream>>>(wq_l,  wq_l_t,  Cc, Cc);
  k_cvt_t<<<512, 256, 0, stream>>>(wkv_l, wkv_l_t, Cc, 2 * Cc);
  k_cvt_t<<<256, 256, 0, stream>>>(wq_g,  wq_g_t,  Cc, Cc);
  k_cvt_t<<<512, 256, 0, stream>>>(wkv_g, wkv_g_t, Cc, 2 * Cc);
  k_cvt_t<<<256, 256, 0, stream>>>(wp,    wp_t,    Cc, Cc);

  // window scores + top-k
  k_topk<<<Bb, 32, 0, stream>>>(unc, idx);

  // gather filtered windows into Acc (the evolving wf, f32)
  k_gather<<<NWSEL, 256, 0, stream>>>(fm, idx, Acc);

  // local window self-attention
  k_qkv_win<0><<<NWSEL, 256, 0, stream>>>(Acc, wq_l_t,  0,  Qb);
  k_qkv_win<0><<<NWSEL, 256, 0, stream>>>(Acc, wkv_l_t, 0,  Kb);
  k_qkv_win<1><<<NWSEL, 256, 0, stream>>>(Acc, wkv_l_t, Cc, Vbt);
  k_win_attn<<<NWSEL, 256, 0, stream>>>(Acc, Qb, Kb, Vbt, WINSZ, 1);

  // global branch: depthwise sr-conv -> LN -> GELU -> KV
  k_conv<<<Bb * Cc, 256, 0, stream>>>(fm, srw, srb, Gtmp);
  k_lngelu<<<Bb * HW1, 256, 0, stream>>>(Gtmp, lng, lnb, Gbf);
  k_kvg<<<(Bb * HW1) / 64, 256, 0, stream>>>(Gbf, wkv_g_t, Kg, Vgt);

  // global cross-attention (filtered windows attend to sr tokens)
  k_qkv_win<0><<<NWSEL, 256, 0, stream>>>(Acc, wq_g_t, 0, Qb);
  k_win_attn<<<NWSEL, 256, 0, stream>>>(Acc, Qb, Kg, Vgt, L_GLOB, 0);

  // assemble output: copy feature map, project + scatter-add windows
  k_copy<<<2048, 256, 0, stream>>>(fm, out, (size_t)Bb * Cc * Hh * Ww / 4);
  k_proj<<<NWSEL, 256, 0, stream>>>(Acc, idx, wp_t, bp, out);
}
